// QLSTM___65481071399193
// MI455X (gfx1250) — matrix operand order repaired and checked by reading, not compiled, not run on hardware
//
#include <hip/hip_runtime.h>
#include <hip/hip_bf16.h>

typedef __attribute__((ext_vector_type(16))) __bf16 v16bf;
typedef __attribute__((ext_vector_type(8)))  float  v8f;

constexpr int SEQ   = 128;
constexpr int BATCH = 1024;
constexpr int IN    = 64;          // input_dim
constexpr int NQ    = 10;          // n qubits / hidden
constexpr int NG    = 4;           // gates
constexpr int D     = IN + NQ;     // 74
constexpr int NCOL  = NG * NQ;     // 40
constexpr int NPAD  = 48;          // N padded to 3 wmma tiles
constexpr int MROWS = SEQ * BATCH; // 131072

// ---- gfx1250 async-to-LDS support (guarded; falls back to direct loads) ----
#if defined(__gfx1250__) && __has_builtin(__builtin_amdgcn_global_load_async_to_lds_b128)
#define HAVE_ASYNC_LDS 1
// Builtin signature (from clang diagnostic): first param is AS1 pointer to
// 'int __attribute__((vector_size(16)))'; args: (global, lds, imm off, imm cpol).
typedef int b128_t __attribute__((vector_size(16)));
typedef __attribute__((address_space(1))) b128_t* gptr_b128;
typedef __attribute__((address_space(3))) b128_t* lptr_b128;
__device__ __forceinline__ void async_copy_b128(const float* g, float* l) {
  __builtin_amdgcn_global_load_async_to_lds_b128(
      (gptr_b128)(g), (lptr_b128)(l), /*offset=*/0, /*cpol=*/0);
}
__device__ __forceinline__ void async_wait0() {
#if __has_builtin(__builtin_amdgcn_s_wait_asynccnt)
  __builtin_amdgcn_s_wait_asynccnt(0);
#else
  asm volatile("s_wait_asynccnt 0x0" ::: "memory");
#endif
}
#else
#define HAVE_ASYNC_LDS 0
#endif

// ---------------------------------------------------------------------------
// Phase 1: preact[t*B+b][g*10+w] = x_t[b,:] . W[g,w,0:64] + (b+theta)[g,w]
// bf16 WMMA GEMM: M=131072 (16-row tiles), N=48 (3 tiles), K=64 (2 k-steps).
// ---------------------------------------------------------------------------
__global__ __launch_bounds__(256)
void qlstm_xproj_wmma(const float* __restrict__ X,    // (SEQ*BATCH, 64)
                      const float* __restrict__ W,    // (4,10,74) row-major
                      const float* __restrict__ Bb,   // (4,10)
                      const float* __restrict__ Th,   // (4,10)
                      float* __restrict__ pre)        // (SEQ*BATCH, 40)
{
  // B matrix staged in LDS as [n][k] (transposed) so each lane's 16-element
  // bf16 fragment is one contiguous 32-byte LDS read.
  __shared__ __align__(32) __bf16 Bs[NPAD * IN];  // 48*64 bf16 = 6 KB
  __shared__ float bias_s[NPAD];

  const int tid = threadIdx.x;
  for (int idx = tid; idx < NPAD * IN; idx += 256) {
    const int n = idx >> 6;          // col = g*10+w
    const int k = idx & 63;          // input dim
    const float v = (n < NCOL) ? W[n * D + k] : 0.0f;
    Bs[idx] = (__bf16)v;
  }
  if (tid < NPAD) bias_s[tid] = (tid < NCOL) ? (Bb[tid] + Th[tid]) : 0.0f;
  __syncthreads();

  const int lane  = tid & 31;
  const int wid   = tid >> 5;
  const int half  = lane >> 4;       // lane half selects K sub-range
  const int nl    = lane & 15;
  const int mtile = blockIdx.x * 8 + wid;

  // A fragment: lane holds row (mtile*16 + nl); contiguous K layout
  // matching B: element e -> K = kc*32 + half*16 + e.
  const long mrow = (long)mtile * 16 + nl;
  const float* arow = X + mrow * IN;

  v16bf afrag[2];
#pragma unroll
  for (int kc = 0; kc < 2; ++kc) {
    const float* p0 = arow + kc * 32 + half * 16;
#pragma unroll
    for (int e = 0; e < 8; ++e) {
      afrag[kc][e]     = (__bf16)p0[e];
      afrag[kc][e + 8] = (__bf16)p0[8 + e];
    }
  }

#pragma unroll
  for (int n0 = 0; n0 < NPAD; n0 += 16) {
    const int col = n0 + nl;
    // B fragment: lane holds column `col`; lanes 0-15 K=kc*32+0..15,
    // lanes 16-31 K=kc*32+16..31 -> contiguous in the [n][k] LDS layout.
    const v16bf b0 = *(const v16bf*)&Bs[col * IN + 0  + half * 16];
    const v16bf b1 = *(const v16bf*)&Bs[col * IN + 32 + half * 16];

    const float bv = bias_s[col];
    v8f c;
#pragma unroll
    for (int e = 0; e < 8; ++e) c[e] = bv;   // accumulate on top of bias

    c = __builtin_amdgcn_wmma_f32_16x16x32_bf16(false, afrag[0], false, b0,
                                                (short)0, c, false, false);
    c = __builtin_amdgcn_wmma_f32_16x16x32_bf16(false, afrag[1], false, b1,
                                                (short)0, c, false, false);

    if (col < NCOL) {
#pragma unroll
      for (int v = 0; v < 8; ++v) {          // C layout: M = v + 8*half
        const long row = (long)mtile * 16 + v + 8 * half;
        pre[row * NCOL + col] = c[v];
      }
    }
  }
}

// ---------------------------------------------------------------------------
// Phase 2: sequential scan. Each thread owns one (batch-row, gate) pair:
// angles = preact + hx . Wh, qlayer == cos + prefix-products (closed form),
// then gate activations; LDS exchange combines gates into cx/hx update.
// Preactivation tiles are double-buffered into LDS with async-to-LDS so the
// global-load latency is hidden behind each step's compute.
// ---------------------------------------------------------------------------
__global__ __launch_bounds__(64)
void qlstm_recurrent(const float* __restrict__ pre,  // (SEQ*BATCH, 40)
                     const float* __restrict__ W,    // (4,10,74)
                     float* __restrict__ out)        // outputs | hx | cx
{
  __shared__ float wh[NG * NQ * NQ];   // [g*10+w][j] recurrent weights
  __shared__ float hx[16 * NQ];
  __shared__ float cxs[16 * NQ];
  __shared__ float act[NG][16 * NQ];
#if HAVE_ASYNC_LDS
  __shared__ __align__(16) float pbuf[2][16 * NCOL]; // double-buffered preact tile
#endif

  const int tid = threadIdx.x;
  const int bl  = tid & 15;            // local batch row
  const int g   = tid >> 4;            // gate
  const int b   = blockIdx.x * 16 + bl;

  for (int idx = tid; idx < NG * NQ * NQ; idx += 64) {
    const int n = idx / NQ;            // g*10+w
    const int j = idx % NQ;
    wh[idx] = W[n * D + IN + j];
  }
  for (int idx = tid; idx < 16 * NQ; idx += 64) { hx[idx] = 0.0f; cxs[idx] = 0.0f; }

#if HAVE_ASYNC_LDS
  {  // prologue: stage t = 0 tile (16 rows x 40 cols = 160 contiguous b128)
    const float* gsrc = pre + (long)blockIdx.x * 16 * NCOL;
#pragma unroll
    for (int i = 0; i < 3; ++i) {
      const int cidx = tid + i * 64;
      if (cidx < 160) async_copy_b128(gsrc + cidx * 4, &pbuf[0][cidx * 4]);
    }
    async_wait0();
  }
#endif
  __syncthreads();

  for (int t = 0; t < SEQ; ++t) {
#if HAVE_ASYNC_LDS
    const int cur = t & 1;
    if (t + 1 < SEQ) {  // stage next timestep while this one computes
      const float* gsrc = pre + ((long)(t + 1) * BATCH + (long)blockIdx.x * 16) * NCOL;
#pragma unroll
      for (int i = 0; i < 3; ++i) {
        const int cidx = tid + i * 64;
        if (cidx < 160) async_copy_b128(gsrc + cidx * 4, &pbuf[cur ^ 1][cidx * 4]);
      }
    }
    const float* prow = &pbuf[cur][bl * NCOL + g * NQ];
#else
    const float* prow = pre + ((long)t * BATCH + b) * NCOL + g * NQ;
    __builtin_prefetch(prow + BATCH * NCOL, 0, 1);   // next timestep
#endif

    float h[NQ];
#pragma unroll
    for (int j = 0; j < NQ; ++j) h[j] = hx[bl * NQ + j];

    float c[NQ];
#pragma unroll
    for (int w = 0; w < NQ; ++w) {
      float a = prow[w];
      const float* whw = &wh[(g * NQ + w) * NQ];
#pragma unroll
      for (int j = 0; j < NQ; ++j) a = __builtin_fmaf(h[j], whw[j], a);
      c[w] = __cosf(a);
    }

    // Closed-form qlayer: m[w>=1] = prod_{j<=w} c_j ; m[0] = prod_{j>=1} c_j
    float mv[NQ];
    float run = c[0];
#pragma unroll
    for (int w = 1; w < NQ; ++w) { run *= c[w]; mv[w] = run; }
    float p0 = c[1];
#pragma unroll
    for (int w = 2; w < NQ; ++w) p0 *= c[w];
    mv[0] = p0;

#pragma unroll
    for (int w = 0; w < NQ; ++w) {
      const float x = mv[w];
      float r;
      if (g == 2) { const float e = __expf(2.0f * x); r = (e - 1.0f) / (e + 1.0f); } // tanh
      else        { r = 1.0f / (1.0f + __expf(-x)); }                                // sigmoid
      act[g][bl * NQ + w] = r;
    }
    __syncthreads();

#pragma unroll
    for (int q = 0; q < 3; ++q) {
      const int idx = tid + q * 64;
      if (idx < 16 * NQ) {
        const float f  = act[0][idx];
        const float i  = act[1][idx];
        const float gg = act[2][idx];
        const float o  = act[3][idx];
        const float c2 = __builtin_fmaf(f, cxs[idx], i * gg);
        const float e  = __expf(2.0f * c2);
        const float h2 = o * ((e - 1.0f) / (e + 1.0f));
        cxs[idx] = c2;
        hx[idx]  = h2;
        const int r = idx / NQ, w = idx % NQ;
        out[((long)t * BATCH + (long)blockIdx.x * 16 + r) * NQ + w] = h2;
      }
    }
#if HAVE_ASYNC_LDS
    async_wait0();        // next tile landed (latency hidden behind this step)
#endif
    __syncthreads();
  }

  // Final hx, cx appended after outputs (tuple return order).
#pragma unroll
  for (int q = 0; q < 3; ++q) {
    const int idx = tid + q * 64;
    if (idx < 16 * NQ) {
      const int r = idx / NQ, w = idx % NQ;
      const long bb = (long)blockIdx.x * 16 + r;
      out[(long)SEQ * BATCH * NQ + bb * NQ + w] = hx[idx];
      out[(long)SEQ * BATCH * NQ + (long)BATCH * NQ + bb * NQ + w] = cxs[idx];
    }
  }
}

// ---------------------------------------------------------------------------
extern "C" void kernel_launch(void* const* d_in, const int* in_sizes, int n_in,
                              void* d_out, int out_size, void* d_ws, size_t ws_size,
                              hipStream_t stream)
{
  const float* X  = (const float*)d_in[0];  // (128,1024,64)
  const float* W  = (const float*)d_in[1];  // (4,10,74)
  const float* Bb = (const float*)d_in[2];  // (4,10)
  const float* Th = (const float*)d_in[3];  // (4,10)
  float* out = (float*)d_out;
  float* pre = (float*)d_ws;                // needs 131072*40*4 = 21.0 MB

  // Phase 1: 8192 M-tiles, 8 tiles (waves) per 256-thread block.
  qlstm_xproj_wmma<<<MROWS / 16 / 8, 256, 0, stream>>>(X, W, Bb, Th, pre);
  // Phase 2: 64 blocks x 64 threads, 16 batch rows per block.
  qlstm_recurrent<<<BATCH / 16, 64, 0, stream>>>(pre, W, out);
}
